// Retention_37357625540634
// MI455X (gfx1250) — matrix operand order repaired and checked by reading, not compiled, not run on hardware
//
#include <hip/hip_runtime.h>
#include <math.h>
#include <stdint.h>

// Problem constants
#define B_ 2
#define N_ 2048
#define H_ 16
#define D_ 64
#define C_ 1024
#define K3_ 3072

typedef __attribute__((ext_vector_type(16))) _Float16 v16h;
typedef __attribute__((ext_vector_type(8)))  _Float16 v8h;
typedef __attribute__((ext_vector_type(8)))  float    v8f;

static __device__ __forceinline__ v8f wmma32(v16h a, v16h b, v8f c) {
  // D = A(16x32 f16) * B(32x16 f16) + C(16x16 f32)
  return __builtin_amdgcn_wmma_f32_16x16x32_f16(false, a, false, b, (short)0, c,
                                                false, false);
}

// Async DMA: copy 16 bytes global -> LDS without touching VGPR data paths.
// lds_off = low 32 bits of flat shared address (== DS byte address).
static __device__ __forceinline__ void async_cp16(uint32_t lds_off, const void* gp) {
  asm volatile("global_load_async_to_lds_b128 %0, %1, off"
               :: "v"(lds_off), "v"(gp) : "memory");
}
static __device__ __forceinline__ void wait_async0() {
  asm volatile("s_wait_asynccnt 0x0" ::: "memory");
}
static __device__ __forceinline__ uint32_t lds_addr(const void* p) {
  return (uint32_t)(uintptr_t)p;
}

// A fragment (16x32, f16), tile stored row-major with K contiguous.
// ISA layout: lanes 0-15: a[0..7]=K0..7, a[8..15]=K16..23 (M=lane)
//             lanes 16-31: a[0..7]=K8..15, a[8..15]=K24..31 (M=lane-16)
static __device__ __forceinline__ v16h fragA(const _Float16* p, int ld, int lane) {
  const int m  = lane & 15;
  const int kb = (lane & 16) ? 8 : 0;
  const _Float16* r = p + m * ld;
  v8h lo = *(const v8h*)(r + kb);
  v8h hi = *(const v8h*)(r + 16 + kb);
  v16h a;
#pragma unroll
  for (int i = 0; i < 8; ++i) { a[i] = lo[i]; a[8 + i] = hi[i]; }
  return a;
}

// B fragment (32x16, f16), tile stored column-major (per output column, K contiguous).
// ISA layout: lanes 0-15: b[0..7]=K0..7, b[8..15]=K16..23 (N=lane); lanes 16-31: b[0..7]=K8..15, b[8..15]=K24..31.
static __device__ __forceinline__ v16h fragB(const _Float16* p, int ld, int lane) {
  const int n  = lane & 15;
  const int kb = (lane & 16) ? 8 : 0;
  const _Float16* r = p + n * ld + kb;
  v8h lo = *(const v8h*)(r);
  v8h hi = *(const v8h*)(r + 16);
  v16h b;
#pragma unroll
  for (int i = 0; i < 8; ++i) { b[i] = lo[i]; b[8 + i] = hi[i]; }
  return b;
}

// ---------------------------------------------------------------------------
// Kernel 1: qkv = x @ W_qkv + b_qkv; split into q(scaled), k, v  f16 [B,H,N,D]
// ---------------------------------------------------------------------------
__global__ __launch_bounds__(128) void qkv_gemm(
    const float* __restrict__ X, const float* __restrict__ W,
    const float* __restrict__ bias,
    _Float16* __restrict__ qb, _Float16* __restrict__ kb, _Float16* __restrict__ vb)
{
  __shared__ alignas(16) _Float16 As[64][40];   // [row][k]
  __shared__ alignas(16) _Float16 Wt[64][40];   // [col][k] (transposed)
  const int tid  = threadIdx.x;
  const int lane = tid & 31;
  const int wave = tid >> 5;
  const int m0 = blockIdx.y * 64;
  const int n0 = blockIdx.x * 64;

  v8f acc[4] = {};
  for (int k0 = 0; k0 < C_; k0 += 32) {
    for (int i = tid; i < 64 * 32; i += 128) {
      int r = i >> 5, c = i & 31;
      As[r][c] = (_Float16)X[(size_t)(m0 + r) * C_ + k0 + c];
    }
    for (int i = tid; i < 32 * 64; i += 128) {
      int c = i & 63, r = i >> 6;
      Wt[c][r] = (_Float16)W[(size_t)(k0 + r) * K3_ + n0 + c];
    }
    __syncthreads();
    v16h a = fragA(&As[wave * 16][0], 40, lane);
#pragma unroll
    for (int t = 0; t < 4; ++t) {
      v16h b = fragB(&Wt[t * 16][0], 40, lane);
      acc[t] = wmma32(a, b, acc[t]);
    }
    __syncthreads();
  }

  const int rowoff = wave * 16 + ((lane & 16) ? 8 : 0);
  const int cb = lane & 15;
#pragma unroll
  for (int t = 0; t < 4; ++t) {
#pragma unroll
    for (int r = 0; r < 8; ++r) {
      int m = m0 + rowoff + r;          // flat (b*N + n)
      int j = n0 + t * 16 + cb;         // [0, 3C)
      float v = acc[t][r] + bias[j];
      int s = j >> 10;                  // 0=q 1=k 2=v
      int h = (j >> 6) & 15;
      int d = j & 63;
      int b = m >> 11;                  // / N
      int nq = m & (N_ - 1);
      size_t idx = ((size_t)(b * H_ + h) * N_ + nq) * D_ + d;
      if (s == 0)      qb[idx] = (_Float16)(v * 0.125f);  // D^-0.5
      else if (s == 1) kb[idx] = (_Float16)v;
      else             vb[idx] = (_Float16)v;
    }
  }
}

// ---------------------------------------------------------------------------
// Kernel 2: streaming retention per (b,h): ret = (q k^T * mask) v + cross
// Grid: (N/64, B*H). Block 128 thr. Never materializes N x N.
// ---------------------------------------------------------------------------
__global__ __launch_bounds__(128) void retention_kernel(
    const _Float16* __restrict__ qb, const _Float16* __restrict__ kb,
    const _Float16* __restrict__ vb, const float* __restrict__ mask,
    const float* __restrict__ gamma, const float* __restrict__ state_prev,
    _Float16* __restrict__ retbuf)
{
  __shared__ alignas(16) _Float16 qs[64][72];    // [row][d]
  __shared__ alignas(16) _Float16 ks[64][72];    // [key][d]
  __shared__ alignas(16) _Float16 vst[64][72];   // [d][key] (transposed)
  __shared__ alignas(16) _Float16 ss[64][72];    // [row][key] masked scores
  __shared__ alignas(16) _Float16 spt[64][72];   // [e][d]  state_prev^T

  const int tid  = threadIdx.x;
  const int lane = tid & 31;
  const int wave = tid >> 5;
  const int bh = blockIdx.y;
  const int b  = bh >> 4;
  const int h  = bh & 15;
  const int m0 = blockIdx.x * 64;

  const _Float16* qbase = qb + (size_t)bh * N_ * D_;
  const _Float16* kbase = kb + (size_t)bh * N_ * D_;
  const _Float16* vbase = vb + (size_t)bh * N_ * D_;
  const float*    spb   = state_prev + (size_t)bh * D_ * D_;
  const float g = gamma[h];

  // q tile: pure f16 copy -> async DMA to LDS (4 x b128 per thread)
  for (int i = tid; i < 512; i += 128) {
    int r = i >> 3, c8 = i & 7;
    async_cp16(lds_addr(&qs[r][c8 * 8]), qbase + (size_t)(m0 + r) * D_ + c8 * 8);
  }
  // state_prev^T needs transpose+convert: VALU path
  for (int i = tid; i < 64 * 64; i += 128) {
    int r = i >> 6, d = i & 63;
    spt[d][r] = (_Float16)spb[r * D_ + d];       // spt[e][d] = sp[d][e]
  }

  v8f ret[4] = {};
  const int rowoff = wave * 16 + ((lane & 16) ? 8 : 0);
  const int cb = lane & 15;
  const float* maskrow = mask + ((size_t)h * N_ + m0) * N_;

  for (int j0 = 0; j0 < N_; j0 += 64) {
    // k tile: pure f16 copy -> async DMA to LDS
    for (int i = tid; i < 512; i += 128) {
      int r = i >> 3, c8 = i & 7;
      async_cp16(lds_addr(&ks[r][c8 * 8]), kbase + (size_t)(j0 + r) * D_ + c8 * 8);
    }
    // v tile transposed: VALU path
    for (int i = tid; i < 64 * 64; i += 128) {
      int key = i >> 6, d = i & 63;
      vst[d][key] = vbase[(size_t)(j0 + key) * D_ + d];
    }
    // prefetch next iteration's mask tile while we compute on this one
    if (j0 + 64 < N_) {
      int pr = tid >> 1, ph = tid & 1;
      __builtin_prefetch(maskrow + (size_t)pr * N_ + (j0 + 64) + ph * 32, 0, 0);
    }
    wait_async0();
    __syncthreads();

    // S = q k^T   (cols = keys, K = d)
    v8f s[4] = {};
#pragma unroll
    for (int kk = 0; kk < 64; kk += 32) {
      v16h a = fragA(&qs[wave * 16][kk], 72, lane);
#pragma unroll
      for (int t = 0; t < 4; ++t) {
        v16h bf = fragB(&ks[t * 16][kk], 72, lane);
        s[t] = wmma32(a, bf, s[t]);
      }
    }
    // mask and park in LDS (C layout -> A layout crossing lanes)
#pragma unroll
    for (int t = 0; t < 4; ++t) {
#pragma unroll
      for (int r = 0; r < 8; ++r) {
        int row = rowoff + r;
        int col = t * 16 + cb;
        float mv = maskrow[(size_t)row * N_ + (j0 + col)];
        ss[row][col] = (_Float16)(s[t][r] * mv);
      }
    }
    __syncthreads();

    // ret += S @ v   (cols = d, K = keys)
#pragma unroll
    for (int kk = 0; kk < 64; kk += 32) {
      v16h a = fragA(&ss[wave * 16][kk], 72, lane);
#pragma unroll
      for (int t = 0; t < 4; ++t) {
        v16h bf = fragB(&vst[t * 16][kk], 72, lane);
        ret[t] = wmma32(a, bf, ret[t]);
      }
    }
    __syncthreads();
  }

  // cross-chunk: ret += (gamma^(n+1) * q) @ state_prev
  {
    const int m = lane & 15;
    const int kb2 = (lane & 16) ? 8 : 0;
    const int row = wave * 16 + m;
    const float sc = powf(g, (float)(m0 + row + 1));
#pragma unroll
    for (int kk = 0; kk < 64; kk += 32) {
      const _Float16* r = &qs[row][kk];
      v16h a;
#pragma unroll
      for (int i = 0; i < 8; ++i) {
        a[i]     = (_Float16)((float)r[kb2 + i] * sc);
        a[8 + i] = (_Float16)((float)r[16 + kb2 + i] * sc);
      }
#pragma unroll
      for (int t = 0; t < 4; ++t) {
        v16h bf = fragB(&spt[t * 16][kk], 72, lane);
        ret[t] = wmma32(a, bf, ret[t]);
      }
    }
  }

  // write ret as f16 into [B,N,C] for the output GEMM
#pragma unroll
  for (int t = 0; t < 4; ++t) {
#pragma unroll
    for (int r = 0; r < 8; ++r) {
      int row = m0 + rowoff + r;       // n
      int col = t * 16 + cb;           // d
      retbuf[((size_t)b * N_ + row) * C_ + h * D_ + col] = (_Float16)ret[t][r];
    }
  }
}

// ---------------------------------------------------------------------------
// Kernel 3: state = sum_n gamma^(N-1-n) k_n^T v_n + gamma^N * state_prev
// One block per (b,h). 64x64 output, K = N = 2048.
// ---------------------------------------------------------------------------
__global__ __launch_bounds__(128) void state_kernel(
    const _Float16* __restrict__ kb, const _Float16* __restrict__ vb,
    const float* __restrict__ gamma, const float* __restrict__ state_prev,
    float* __restrict__ out_state)
{
  __shared__ alignas(16) _Float16 kswt[64][40];  // [d][n_local], decay-weighted
  __shared__ alignas(16) _Float16 vtt[64][40];   // [e][n_local]
  const int tid  = threadIdx.x;
  const int lane = tid & 31;
  const int wave = tid >> 5;
  const int bh = blockIdx.x;
  const int h  = bh & 15;
  const float g  = gamma[h];
  const float lg = log2f(g);

  const _Float16* kbase = kb + (size_t)bh * N_ * D_;
  const _Float16* vbase = vb + (size_t)bh * N_ * D_;

  v8f acc[4] = {};
  for (int n0 = 0; n0 < N_; n0 += 32) {
    for (int i = tid; i < 32 * 64; i += 128) {
      int n = i >> 6, d = i & 63;
      int ng = n0 + n;
      float w = exp2f(lg * (float)(N_ - 1 - ng));   // gamma^(N-1-n)
      kswt[d][n] = (_Float16)((float)kbase[(size_t)ng * D_ + d] * w);
      vtt[d][n]  = vbase[(size_t)ng * D_ + d];
    }
    __syncthreads();
    v16h a = fragA(&kswt[wave * 16][0], 40, lane);
#pragma unroll
    for (int t = 0; t < 4; ++t) {
      v16h bf = fragB(&vtt[t * 16][0], 40, lane);
      acc[t] = wmma32(a, bf, acc[t]);
    }
    __syncthreads();
  }

  const float gN = powf(g, (float)N_);
  const int rowoff = wave * 16 + ((lane & 16) ? 8 : 0);
  const int cb = lane & 15;
#pragma unroll
  for (int t = 0; t < 4; ++t) {
#pragma unroll
    for (int r = 0; r < 8; ++r) {
      int d = rowoff + r;
      int e = t * 16 + cb;
      size_t idx = (size_t)bh * D_ * D_ + (size_t)d * D_ + e;
      out_state[idx] = acc[t][r] + state_prev[idx] * gN;
    }
  }
}

// ---------------------------------------------------------------------------
// Kernel 4: out = ret @ W_out + b_out   (4096 x 1024 x 1024), f32 out
// ---------------------------------------------------------------------------
__global__ __launch_bounds__(128) void out_gemm(
    const _Float16* __restrict__ A, const float* __restrict__ W,
    const float* __restrict__ bias, float* __restrict__ out)
{
  __shared__ alignas(16) _Float16 As[64][40];
  __shared__ alignas(16) _Float16 Wt[64][40];
  const int tid  = threadIdx.x;
  const int lane = tid & 31;
  const int wave = tid >> 5;
  const int m0 = blockIdx.y * 64;
  const int n0 = blockIdx.x * 64;

  v8f acc[4] = {};
  for (int k0 = 0; k0 < C_; k0 += 32) {
    // A tile: pure f16 copy -> async DMA to LDS (2 x b128 per thread)
    for (int i = tid; i < 256; i += 128) {
      int r = i >> 2, c8 = i & 3;
      async_cp16(lds_addr(&As[r][c8 * 8]),
                 A + (size_t)(m0 + r) * C_ + k0 + c8 * 8);
    }
    for (int i = tid; i < 32 * 64; i += 128) {
      int c = i & 63, r = i >> 6;
      Wt[c][r] = (_Float16)W[(size_t)(k0 + r) * C_ + n0 + c];
    }
    wait_async0();
    __syncthreads();
    v16h a = fragA(&As[wave * 16][0], 40, lane);
#pragma unroll
    for (int t = 0; t < 4; ++t) {
      v16h b = fragB(&Wt[t * 16][0], 40, lane);
      acc[t] = wmma32(a, b, acc[t]);
    }
    __syncthreads();
  }

  const int rowoff = wave * 16 + ((lane & 16) ? 8 : 0);
  const int cb = lane & 15;
#pragma unroll
  for (int t = 0; t < 4; ++t) {
#pragma unroll
    for (int r = 0; r < 8; ++r) {
      int m = m0 + rowoff + r;
      int n = n0 + t * 16 + cb;
      out[(size_t)m * C_ + n] = acc[t][r] + bias[n];
    }
  }
}

// ---------------------------------------------------------------------------
extern "C" void kernel_launch(void* const* d_in, const int* in_sizes, int n_in,
                              void* d_out, int out_size, void* d_ws, size_t ws_size,
                              hipStream_t stream) {
  (void)in_sizes; (void)n_in; (void)out_size; (void)ws_size;
  const float* x          = (const float*)d_in[0];
  const float* mask       = (const float*)d_in[1];
  const float* gamma      = (const float*)d_in[2];
  const float* state_prev = (const float*)d_in[3];
  const float* W_qkv      = (const float*)d_in[4];
  const float* b_qkv      = (const float*)d_in[5];
  const float* W_out      = (const float*)d_in[6];
  const float* b_out      = (const float*)d_in[7];

  float* out       = (float*)d_out;                       // [B,N,C]
  float* out_state = out + (size_t)B_ * N_ * C_;          // [B,H,D,D]

  const size_t ELT = (size_t)B_ * H_ * N_ * D_;           // 4,194,304
  _Float16* qb = (_Float16*)d_ws;
  _Float16* kb = qb + ELT;
  _Float16* vb = kb + ELT;
  _Float16* rb = vb + ELT;                                // ret f16 [B,N,C]

  // 1) QKV projection + split
  qkv_gemm<<<dim3(K3_ / 64, (B_ * N_) / 64), 128, 0, stream>>>(
      x, W_qkv, b_qkv, qb, kb, vb);

  // 2) Streaming masked retention + cross-chunk term
  retention_kernel<<<dim3(N_ / 64, B_ * H_), 128, 0, stream>>>(
      qb, kb, vb, mask, gamma, state_prev, rb);

  // 3) Recurrent state update
  state_kernel<<<dim3(B_ * H_), 128, 0, stream>>>(
      kb, vb, gamma, state_prev, out_state);

  // 4) Output projection
  out_gemm<<<dim3(C_ / 64, (B_ * N_) / 64), 128, 0, stream>>>(
      rb, W_out, b_out, out);
}
